// SimSceLayer_53961969107141
// MI455X (gfx1250) — matrix operand order repaired and checked by reading, not compiled, not run on hardware
//
#include <hip/hip_runtime.h>
#include <hip/hip_bf16.h>

// ---------------------------------------------------------------------------
// SimCSE contrastive loss for MI455X (gfx1250, wave32, WMMA)
//   feat  = concat(f1, f2)             : (8192, 1024) f32
//   n     = feat / max(||feat||,1e-8)  -> f16 in ws (16 MB, L2-resident)
//   S     = n @ n^T                    : v_wmma_f32_16x16x32_f16, f32 accum
//   logits= S * 20 - eye*1e10          (fused epilogue, NT stores)
//   loss  = weighted CE against y_true
// d_out = [loss(1), logits(8192*8192), y_true(8192)] as float
// ---------------------------------------------------------------------------

typedef _Float16 half16 __attribute__((ext_vector_type(16)));
typedef _Float16 half8  __attribute__((ext_vector_type(8)));
typedef float    float8 __attribute__((ext_vector_type(8)));

#define DIM      1024
#define BS       8192      // 2*b
#define HALF_BS  4096
#define TAU_INV  20.0f
#define BIGPOS   1.0e10f
#define COS_EPS  1e-8f

// ---------------------------------------------------------------- init ------
__global__ void init_kernel(float* accum) {
    if (threadIdx.x < 2) accum[threadIdx.x] = 0.0f;
}

// ----------------------------------------------------------- normalize ------
// One block per row: sum of squares -> clamped norm -> f16 normalized row.
__global__ void __launch_bounds__(256)
normalize_kernel(const float* __restrict__ f1, const float* __restrict__ f2,
                 _Float16* __restrict__ nrm) {
    const int row = blockIdx.x;
    const float* src = (row < HALF_BS) ? (f1 + (size_t)row * DIM)
                                       : (f2 + (size_t)(row - HALF_BS) * DIM);
    __shared__ float red[8];

    float ss = 0.0f;
    for (int k = threadIdx.x; k < DIM; k += 256) {
        float v = src[k];
        ss += v * v;
    }
    #pragma unroll
    for (int off = 16; off > 0; off >>= 1) ss += __shfl_down(ss, off, 32);
    if ((threadIdx.x & 31) == 0) red[threadIdx.x >> 5] = ss;
    __syncthreads();
    if (threadIdx.x < 8) {
        float v = red[threadIdx.x];
        #pragma unroll
        for (int off = 4; off > 0; off >>= 1) v += __shfl_down(v, off, 8);
        if (threadIdx.x == 0) red[0] = v;
    }
    __syncthreads();

    const float norm = fmaxf(sqrtf(red[0]), COS_EPS);
    const float inv  = 1.0f / norm;
    for (int k = threadIdx.x; k < DIM; k += 256)
        nrm[(size_t)row * DIM + k] = (_Float16)(src[k] * inv);
}

// ---------------------------------------------------------------- GEMM ------
// C(8192x8192) = Nrm * Nrm^T, fused *20 and diagonal -1e10.
// Block = 256 threads = 8 waves arranged 2(M) x 4(N).
// Wave tile = 64x64 = 4x4 WMMA tiles  -> 16 wmma per 16 b128 loads per k-step.
// Block tile = 128(M) x 256(N).
//
// CDNA5 wave32 fragment layouts (ISA 7.12.2):
//   A 16x32 f16 : lane&15 = row M; element j of v16h is
//                 K = hs*16 + j  (hs = lane>>4; this code's order, same as B)
//   B 32x16 f16 : lane&15 = col N; element j is K = hs*16 + j (this code's order, not the ISA table)
//   C/D 16x16 f32: element v is (M = v + hs*8, N = lane&15)
__global__ void __launch_bounds__(256)
simgemm_wmma(const _Float16* __restrict__ Nrm, float* __restrict__ out) {
    const int lane = threadIdx.x & 31;
    const int wid  = threadIdx.x >> 5;
    const int lm   = lane & 15;   // M/N lane index
    const int hs   = lane >> 4;   // half select

    const int wave_m = wid & 1;   // 0..1
    const int wave_n = wid >> 1;  // 0..3

    const int m0 = blockIdx.y * 128 + wave_m * 64;
    const int n0 = blockIdx.x * 256 + wave_n * 64;

    const _Float16* arow[4];
    const _Float16* brow[4];
    #pragma unroll
    for (int i = 0; i < 4; ++i) {
        arow[i] = Nrm + (size_t)(m0 + 16 * i + lm) * DIM;
        brow[i] = Nrm + (size_t)(n0 + 16 * i + lm) * DIM;
    }

    const int aoff = hs * 16;   // A: K offset of this half-wave
    const int boff = hs * 16;   // B: K offset of this half-wave

    float8 acc[4][4];
    #pragma unroll
    for (int i = 0; i < 4; ++i)
        #pragma unroll
        for (int j = 0; j < 4; ++j)
            acc[i][j] = (float8){};

    for (int k = 0; k < DIM; k += 32) {
        union { half16 v; half8 h[2]; } a[4], b[4];
        #pragma unroll
        for (int i = 0; i < 4; ++i) {
            a[i].v = *(const half16*)(arow[i] + k + aoff);
        }
        #pragma unroll
        for (int j = 0; j < 4; ++j)
            b[j].v = *(const half16*)(brow[j] + k + boff);

        #pragma unroll
        for (int i = 0; i < 4; ++i)
            #pragma unroll
            for (int j = 0; j < 4; ++j)
                acc[i][j] = __builtin_amdgcn_wmma_f32_16x16x32_f16(
                    false, a[i].v, false, b[j].v, (short)0, acc[i][j],
                    false, false);
    }

    // Epilogue: scale by 1/tau, mask diagonal, non-temporal stores so the
    // 256 MB logits stream does not evict the 16 MB operand from L2.
    #pragma unroll
    for (int i = 0; i < 4; ++i) {
        #pragma unroll
        for (int j = 0; j < 4; ++j) {
            const int col = n0 + 16 * j + lm;
            #pragma unroll
            for (int v = 0; v < 8; ++v) {
                const int row = m0 + 16 * i + hs * 8 + v;
                float val = acc[i][j][v] * TAU_INV;
                if (row == col) val -= BIGPOS;
                __builtin_nontemporal_store(val, &out[(size_t)row * BS + col]);
            }
        }
    }
}

// ------------------------------------------------------------ row loss ------
// One block per row: logsumexp over 8192 logits, weighted CE contribution.
__global__ void __launch_bounds__(256)
row_loss_kernel(const float* __restrict__ logits, const int* __restrict__ mask,
                float* __restrict__ accum, float* __restrict__ ytrue_out) {
    const int row = blockIdx.x;
    const float* lrow = logits + (size_t)row * BS;

    __shared__ float redmax[8];
    __shared__ float redsum[8];
    __shared__ float srowmax;

    float mx = -__builtin_inff();
    for (int j = threadIdx.x; j < BS; j += 256) mx = fmaxf(mx, lrow[j]);
    #pragma unroll
    for (int off = 16; off > 0; off >>= 1) mx = fmaxf(mx, __shfl_down(mx, off, 32));
    if ((threadIdx.x & 31) == 0) redmax[threadIdx.x >> 5] = mx;
    __syncthreads();
    if (threadIdx.x < 8) {
        float v = redmax[threadIdx.x];
        #pragma unroll
        for (int off = 4; off > 0; off >>= 1) v = fmaxf(v, __shfl_down(v, off, 8));
        if (threadIdx.x == 0) srowmax = v;
    }
    __syncthreads();
    const float rowmax = srowmax;

    float se = 0.0f;
    for (int j = threadIdx.x; j < BS; j += 256) se += __expf(lrow[j] - rowmax);
    #pragma unroll
    for (int off = 16; off > 0; off >>= 1) se += __shfl_down(se, off, 32);
    if ((threadIdx.x & 31) == 0) redsum[threadIdx.x >> 5] = se;
    __syncthreads();
    if (threadIdx.x == 0) {
        float s = 0.0f;
        #pragma unroll
        for (int w = 0; w < 8; ++w) s += redsum[w];

        const int yt = (row < HALF_BS) ? (row + HALF_BS) : (row - HALF_BS);
        const float w = 1.0f - (float)mask[yt & (HALF_BS - 1)];
        const float lse = logf(s) + rowmax;
        const float per_example = lse - lrow[yt];
        atomicAdd(&accum[0], w * per_example);
        atomicAdd(&accum[1], w);
        ytrue_out[row] = (float)yt;
    }
}

// ------------------------------------------------------------ finalize ------
__global__ void finalize_kernel(const float* accum, float* out) {
    out[0] = accum[0] / accum[1];
}

// ------------------------------------------------------------- launch -------
extern "C" void kernel_launch(void* const* d_in, const int* in_sizes, int n_in,
                              void* d_out, int out_size, void* d_ws, size_t ws_size,
                              hipStream_t stream) {
    (void)in_sizes; (void)n_in; (void)out_size; (void)ws_size;

    const float* f1   = (const float*)d_in[0];
    const float* f2   = (const float*)d_in[1];
    const int*   mask = (const int*)d_in[2];

    float* out    = (float*)d_out;
    float* logits = out + 1;
    float* ytrue  = out + 1 + (size_t)BS * BS;

    float*     accum = (float*)d_ws;                         // 2 floats
    _Float16*  nrm   = (_Float16*)((char*)d_ws + 256);       // 16 MB f16 matrix

    init_kernel<<<1, 64, 0, stream>>>(accum);
    normalize_kernel<<<BS, 256, 0, stream>>>(f1, f2, nrm);

    dim3 grid(BS / 256, BS / 128);   // (N tiles, M tiles)
    simgemm_wmma<<<grid, 256, 0, stream>>>(nrm, logits);

    row_loss_kernel<<<BS, 256, 0, stream>>>(logits, mask, accum, ytrue);
    finalize_kernel<<<1, 1, 0, stream>>>(accum, out);
}
